// SelfAttentionLayer_66812511256723
// MI455X (gfx1250) — matrix operand order repaired and checked by reading, not compiled, not run on hardware
//
#include <hip/hip_runtime.h>
#include <hip/hip_bf16.h>

typedef __bf16 bf16;
typedef __attribute__((ext_vector_type(16))) __bf16 v16bf;
typedef __attribute__((ext_vector_type(8)))  __bf16 v8bf;
typedef __attribute__((ext_vector_type(8)))  float  v8f;

#define N_TOK 4096
#define CDIM  256
#define NHEAD 8
#define HD    32
#define NSEG  8
#define NEGBIAS (-10000.0f)

union AOp { v16bf v; v8bf h[2]; };

// A-operand (16x32 bf16, M rows x K): lane = row (lane&15); VGPR halves:
//   elems 0..15 = K[k0 + 16*hi .. +15]
__device__ __forceinline__ v16bf load_a16(const bf16* base, int row0, int ld, int k0, int lane) {
    AOp a;
    int r  = row0 + (lane & 15);
    int hi = (lane >> 4) & 1;
    const bf16* p = base + (size_t)r * ld + k0 + 16 * hi;
    a.h[0] = *(const v8bf*)(p);
    a.h[1] = *(const v8bf*)(p + 8);
    return a.v;
}

// B-operand (32x16 bf16, K x N): lane = col n (lane&15); lanes 0-15 K=k0..k0+15,
// lanes 16-31 K=k0+16..k0+31 -> one contiguous 32B chunk in an n-major layout.
__device__ __forceinline__ v16bf load_b16(const bf16* baseN, int n0, int ld, int k0, int lane) {
    int n  = n0 + (lane & 15);
    int hi = (lane >> 4) & 1;
    return *(const v16bf*)(baseN + (size_t)n * ld + k0 + 16 * hi);
}

__device__ __forceinline__ v8f wmma_bf16(v16bf a, v16bf b, v8f c) {
    return __builtin_amdgcn_wmma_f32_16x16x32_bf16(false, a, false, b, (short)0, c, false, false);
}

__device__ __forceinline__ float rmax16(float x) {
#pragma unroll
    for (int m = 1; m < 16; m <<= 1) x = fmaxf(x, __shfl_xor(x, m, 32));
    return x;
}
__device__ __forceinline__ float rsum16(float x) {
#pragma unroll
    for (int m = 1; m < 16; m <<= 1) x += __shfl_xor(x, m, 32);
    return x;
}
__device__ __forceinline__ int seg_of(const int* cu, int t) {
    int s = 0;
#pragma unroll
    for (int j = 1; j <= NSEG; j++) s += (t >= cu[j]) ? 1 : 0;
    return s;
}

// ---------------- prep: bf16 conversion + weight transposes ----------------
__global__ void prep_kernel(const float* __restrict__ qkv, const float* __restrict__ qpos,
                            const float* __restrict__ Wq, const float* __restrict__ Wkv,
                            const float* __restrict__ Wproj,
                            bf16* __restrict__ aq, bf16* __restrict__ akv,
                            bf16* __restrict__ WqT, bf16* __restrict__ WkvT,
                            bf16* __restrict__ WprojT) {
    int idx = blockIdx.x * blockDim.x + threadIdx.x;
    const int NC = N_TOK * CDIM;
    if (idx < NC) {
        float x = qkv[idx];
        aq[idx]  = (bf16)(x + qpos[idx]);
        akv[idx] = (bf16)x;
        return;
    }
    idx -= NC;
    if (idx < CDIM * CDIM) {                       // WqT[n][k] = Wq[k][n]
        int k = idx & (CDIM - 1), n = idx >> 8;
        WqT[n * CDIM + k] = (bf16)Wq[k * CDIM + n];
        return;
    }
    idx -= CDIM * CDIM;
    if (idx < 2 * CDIM * CDIM) {                   // WkvT[n][k] = Wkv[k][n], n<512
        int k = idx & (CDIM - 1), n = idx >> 8;
        WkvT[n * CDIM + k] = (bf16)Wkv[k * (2 * CDIM) + n];
        return;
    }
    idx -= 2 * CDIM * CDIM;
    if (idx < CDIM * CDIM) {                       // WprojT[n][k] = Wproj[k][n]
        int k = idx & (CDIM - 1), n = idx >> 8;
        WprojT[n * CDIM + k] = (bf16)Wproj[k * CDIM + n];
    }
}

// ---------------- fused QKV projection GEMM (WMMA) ----------------
// Output columns 0..255 -> qb[tok][c], 256..511 -> kb[tok][c], 512..767 -> vt[d][tok]
__global__ void proj_kernel(const bf16* __restrict__ aq, const bf16* __restrict__ akv,
                            const bf16* __restrict__ WqT, const bf16* __restrict__ WkvT,
                            bf16* __restrict__ qb, bf16* __restrict__ kb,
                            bf16* __restrict__ vt) {
    int lane = threadIdx.x & 31;
    int wave = threadIdx.x >> 5;
    int cb   = (blockIdx.x * 8 + wave) * 16;   // 0..767
    int t0   = blockIdx.y * 16;
    const bf16* Ap = (cb < CDIM) ? aq  : akv;
    const bf16* Wp = (cb < CDIM) ? WqT : WkvT;
    int wrow = (cb < CDIM) ? cb : (cb - CDIM);
    v8f acc = {};
#pragma unroll
    for (int k0 = 0; k0 < CDIM; k0 += 32) {
        v16bf a = load_a16(Ap, t0, CDIM, k0, lane);
        v16bf b = load_b16(Wp, wrow, CDIM, k0, lane);
        acc = wmma_bf16(a, b, acc);
    }
    int n = lane & 15, hi = (lane >> 4) & 1;
    if (cb < 2 * CDIM) {
        bf16* dst = (cb < CDIM) ? qb : kb;
        int dcol  = (cb < CDIM) ? cb : (cb - CDIM);
#pragma unroll
        for (int r = 0; r < 8; r++)
            dst[(size_t)(t0 + r + 8 * hi) * CDIM + dcol + n] = (bf16)acc[r];
    } else {
        int d = cb - 2 * CDIM + n;                 // head-dim column, transpose store
        v8bf pk;
#pragma unroll
        for (int r = 0; r < 8; r++) pk[r] = (bf16)acc[r];
        *(v8bf*)(vt + (size_t)d * N_TOK + t0 + 8 * hi) = pk;  // tokens t0+8hi..+7 contiguous
    }
}

// ---------------- segment-masked flash attention (WMMA, online softmax) ----------------
__global__ void attn_kernel(const bf16* __restrict__ qb, const bf16* __restrict__ kb,
                            const bf16* __restrict__ vt, const int* __restrict__ cu_g,
                            bf16* __restrict__ featb) {
    __shared__ bf16 pbuf[4][16 * 48];              // P tile per wave, padded rows (96B, 16B-aligned)
    int lane = threadIdx.x & 31;
    int wave = threadIdx.x >> 5;
    int h    = blockIdx.y;
    int q0   = (blockIdx.x * 4 + wave) * 16;

    int cu[NSEG + 1];
#pragma unroll
    for (int j = 0; j <= NSEG; j++) cu[j] = cu_g[j];

    int n = lane & 15, hi = (lane >> 4) & 1;
    int segq[8];
#pragma unroll
    for (int r = 0; r < 8; r++) segq[r] = seg_of(cu, q0 + r + 8 * hi);
    int kstart = cu[seg_of(cu, q0)];
    int kend   = cu[seg_of(cu, q0 + 15) + 1];

    v16bf qa = load_a16(qb, q0, CDIM, h * HD, lane);
    const float scale = 0.17677669529663688f;      // 1/sqrt(32)

    v8f o0 = {}, o1 = {};
    float m[8], l[8];
#pragma unroll
    for (int r = 0; r < 8; r++) { m[r] = NEGBIAS; l[r] = 0.0f; }

    bf16* pw = pbuf[wave];
    const v8f zero = {};
    for (int kc = (kstart & ~31); kc < kend; kc += 32) {
        // scores: two 16-key tiles, K-dim = hd = 32 -> one WMMA each
        v16bf b0 = load_b16(kb, kc,      CDIM, h * HD, lane);
        v16bf b1 = load_b16(kb, kc + 16, CDIM, h * HD, lane);
        v8f s0 = wmma_bf16(qa, b0, zero);
        v8f s1 = wmma_bf16(qa, b1, zero);

        int sk0 = seg_of(cu, kc + n);
        int sk1 = seg_of(cu, kc + 16 + n);
        float f0[8], f1[8], mnew[8], alpha[8], p0[8], p1[8];
#pragma unroll
        for (int r = 0; r < 8; r++) {
            f0[r] = (segq[r] == sk0) ? s0[r] * scale : NEGBIAS;
            f1[r] = (segq[r] == sk1) ? s1[r] * scale : NEGBIAS;
        }
#pragma unroll
        for (int r = 0; r < 8; r++) {
            float t = rmax16(fmaxf(f0[r], f1[r]));
            mnew[r]  = fmaxf(m[r], t);
            alpha[r] = __expf(m[r] - mnew[r]);
        }
#pragma unroll
        for (int r = 0; r < 8; r++) {
            p0[r] = __expf(f0[r] - mnew[r]);
            p1[r] = __expf(f1[r] - mnew[r]);
        }
#pragma unroll
        for (int r = 0; r < 8; r++) {
            l[r] = l[r] * alpha[r] + rsum16(p0[r] + p1[r]);
            m[r] = mnew[r];
            o0[r] *= alpha[r];
            o1[r] *= alpha[r];
        }
        // D-layout -> A-layout transpose of P through LDS
#pragma unroll
        for (int r = 0; r < 8; r++) {
            int row = r + 8 * hi;
            pw[row * 48 + n]      = (bf16)p0[r];
            pw[row * 48 + 16 + n] = (bf16)p1[r];
        }
        asm volatile("s_wait_dscnt 0" ::: "memory");   // same-wave LDS RAW
        AOp pa;
        {
            const bf16* p = pw + (lane & 15) * 48 + 16 * hi;
            pa.h[0] = *(const v8bf*)(p);
            pa.h[1] = *(const v8bf*)(p + 8);
        }
        // P @ V: B-operand k-dim = tokens (contiguous in vt), two d-column tiles
        v16bf vb0 = load_b16(vt, h * HD,      N_TOK, kc, lane);
        v16bf vb1 = load_b16(vt, h * HD + 16, N_TOK, kc, lane);
        o0 = wmma_bf16(pa.v, vb0, o0);
        o1 = wmma_bf16(pa.v, vb1, o1);
    }
#pragma unroll
    for (int r = 0; r < 8; r++) {
        float inv = 1.0f / l[r];
        size_t row = (size_t)(q0 + r + 8 * hi);
        featb[row * CDIM + h * HD + n]      = (bf16)(o0[r] * inv);
        featb[row * CDIM + h * HD + 16 + n] = (bf16)(o1[r] * inv);
    }
}

// ---------------- output projection GEMM + bias ----------------
__global__ void outproj_kernel(const bf16* __restrict__ featb, const bf16* __restrict__ WprojT,
                               const float* __restrict__ bproj, float* __restrict__ out) {
    int lane = threadIdx.x & 31;
    int wave = threadIdx.x >> 5;
    int cb   = (blockIdx.x * 8 + wave) * 16;
    int t0   = blockIdx.y * 16;
    v8f acc = {};
#pragma unroll
    for (int k0 = 0; k0 < CDIM; k0 += 32) {
        v16bf a = load_a16(featb, t0, CDIM, k0, lane);
        v16bf b = load_b16(WprojT, cb, CDIM, k0, lane);
        acc = wmma_bf16(a, b, acc);
    }
    int n = lane & 15, hi = (lane >> 4) & 1;
    float bias = bproj[cb + n];
#pragma unroll
    for (int r = 0; r < 8; r++)
        out[(size_t)(t0 + r + 8 * hi) * CDIM + cb + n] = acc[r] + bias;
}

extern "C" void kernel_launch(void* const* d_in, const int* in_sizes, int n_in,
                              void* d_out, int out_size, void* d_ws, size_t ws_size,
                              hipStream_t stream) {
    (void)in_sizes; (void)n_in; (void)out_size; (void)ws_size;
    const float* qkv   = (const float*)d_in[0];
    const float* qpos  = (const float*)d_in[1];
    const int*   cu    = (const int*)d_in[2];
    const float* Wq    = (const float*)d_in[4];
    const float* Wkv   = (const float*)d_in[5];
    const float* Wproj = (const float*)d_in[6];
    const float* bproj = (const float*)d_in[7];
    float* out = (float*)d_out;

    char* ws = (char*)d_ws;
    size_t off = 0;
    auto take = [&](size_t bytes) { char* p = ws + off; off += (bytes + 255) & ~(size_t)255; return p; };
    bf16* aq      = (bf16*)take((size_t)N_TOK * CDIM * 2);
    bf16* akv     = (bf16*)take((size_t)N_TOK * CDIM * 2);
    bf16* WqT     = (bf16*)take((size_t)CDIM * CDIM * 2);
    bf16* WkvT    = (bf16*)take((size_t)2 * CDIM * CDIM * 2);
    bf16* WprojT  = (bf16*)take((size_t)CDIM * CDIM * 2);
    bf16* qb      = (bf16*)take((size_t)N_TOK * CDIM * 2);
    bf16* kb      = (bf16*)take((size_t)N_TOK * CDIM * 2);
    bf16* vt      = (bf16*)take((size_t)CDIM * N_TOK * 2);
    bf16* featb   = (bf16*)take((size_t)N_TOK * CDIM * 2);

    int prep_total = N_TOK * CDIM + CDIM * CDIM + 2 * CDIM * CDIM + CDIM * CDIM;
    prep_kernel<<<(prep_total + 255) / 256, 256, 0, stream>>>(
        qkv, qpos, Wq, Wkv, Wproj, aq, akv, WqT, WkvT, WprojT);

    proj_kernel<<<dim3(6, N_TOK / 16), 256, 0, stream>>>(aq, akv, WqT, WkvT, qb, kb, vt);

    attn_kernel<<<dim3(N_TOK / 64, NHEAD), 128, 0, stream>>>(qb, kb, vt, cu, featb);

    outproj_kernel<<<dim3(2, N_TOK / 16), 256, 0, stream>>>(featb, WprojT, bproj, out);
}
